// LlamaAttention_29025388987001
// MI455X (gfx1250) — matrix operand order repaired and checked by reading, not compiled, not run on hardware
//
#include <hip/hip_runtime.h>
#include <hip/hip_bf16.h>

// ---------------- problem constants ----------------
#define BATCH   8
#define QLEN    64
#define HDIM    4096
#define NH      32
#define NKV     8
#define HD      128
#define LCACHE  4096
#define GRP     (NH / NKV)          // 4
#define CQKV    (NH*HD + 2*NKV*HD)  // 6144
#define NEGINF  (-1e9f)
#define SCALE   0.08838834764831845f

typedef __bf16 v16bf __attribute__((ext_vector_type(16)));
typedef float  v8f   __attribute__((ext_vector_type(8)));
typedef float  v4f   __attribute__((ext_vector_type(4)));

static __device__ __forceinline__ v8f wmma_bf16(v16bf a, v16bf b, v8f c) {
    // D(16x16,f32) = A(16x32,bf16) * B(32x16,bf16) + C
    return __builtin_amdgcn_wmma_f32_16x16x32_bf16(false, a, false, b,
                                                   (short)0, c, false, false);
}

// Async global->LDS 16B copy (per active lane), tracked by ASYNCcnt.
static __device__ __forceinline__ void async_ld_b128(unsigned int lds_byte_addr,
                                                     const void* gptr) {
    asm volatile("global_load_async_to_lds_b128 %0, %1, off"
                 :: "v"(lds_byte_addr), "v"(gptr)
                 : "memory");
}
#define WAIT_ASYNC(N) asm volatile("s_wait_asynccnt " #N ::: "memory")

static __device__ __forceinline__ unsigned int lds_lo32(const void* p) {
    return (unsigned int)(size_t)p;   // low 32 bits of LDS aperture addr = LDS offset
}

// =====================================================================
// Kernel 1: fused QKV projection  C[512 x 6144] = X[512x4096] * [Wq|Wk|Wv]
// block = 128 threads (4 waves). Each wave: 16x16 tile; block: 64 rows x 16 cols.
// W tiles streamed into LDS with async b128, double buffered.
// =====================================================================
__global__ __launch_bounds__(128) void gemm_qkv_kernel(
    const float* __restrict__ X,  const float* __restrict__ Wq,
    const float* __restrict__ Wk, const float* __restrict__ Wv,
    float* __restrict__ out) {
    __shared__ float wtile[2][32][16];
    const int tid  = threadIdx.x;
    const int wave = tid >> 5, lane = tid & 31;
    const int half = lane >> 4, lm = lane & 15;
    const int colBase = blockIdx.x * 16;
    const int rowBase = blockIdx.y * 64 + wave * 16;

    const float* Wp; int wstride, wcol0;
    if (colBase < NH*HD)               { Wp = Wq; wstride = NH*HD;  wcol0 = colBase; }
    else if (colBase < NH*HD + NKV*HD) { Wp = Wk; wstride = NKV*HD; wcol0 = colBase - NH*HD; }
    else                               { Wp = Wv; wstride = NKV*HD; wcol0 = colBase - NH*HD - NKV*HD; }

    const int kk = tid >> 3;          // 0..15? no: 128 threads / 32 rows -> tid*4 scheme
    const int e  = tid * 4;
    const int skk = e >> 4, snn = e & 15;
    const float* xrow = X + (size_t)(rowBase + lm) * HDIM;

    auto stageW = [&](int buf, int k0) {
        const float* wr = Wp + (size_t)(k0 + skk) * wstride + wcol0 + snn;
        async_ld_b128(lds_lo32(&wtile[buf][skk][snn]), wr);
    };

    v8f acc; for (int r = 0; r < 8; ++r) acc[r] = 0.f;

    stageW(0, 0);
    for (int k0 = 0; k0 < HDIM; k0 += 32) {
        const int buf = (k0 >> 5) & 1;
        if (k0 + 32 < HDIM) { stageW(buf ^ 1, k0 + 32); WAIT_ASYNC(0x1); }
        else                { WAIT_ASYNC(0x0); }
        __syncthreads();

        const int kb = k0 + half * 8;
        v4f a0 = *(const v4f*)(xrow + kb);
        v4f a1 = *(const v4f*)(xrow + kb + 4);
        v4f a2 = *(const v4f*)(xrow + kb + 16);
        v4f a3 = *(const v4f*)(xrow + kb + 20);
        v16bf a, b;
        #pragma unroll
        for (int j = 0; j < 4; ++j) {
            a[j]    = (__bf16)a0[j];  a[4+j]  = (__bf16)a1[j];
            a[8+j]  = (__bf16)a2[j];  a[12+j] = (__bf16)a3[j];
        }
        #pragma unroll
        for (int i = 0; i < 16; ++i) b[i] = (__bf16)wtile[buf][(i < 8 ? half*8 + i : 8 + half*8 + i)][lm];
        acc = wmma_bf16(a, b, acc);
        __syncthreads();
    }
    #pragma unroll
    for (int r = 0; r < 8; ++r)
        out[(size_t)(rowBase + r + half*8) * CQKV + colBase + lm] = acc[r];
}

// =====================================================================
// Kernel 2: RoPE + bf16 pack.  q->[B,NH,Q,HD], k,v->[B,NKV,Q,HD]
// =====================================================================
__global__ void rope_kernel(const float* __restrict__ qkv,
                            const float* __restrict__ cosT, const float* __restrict__ sinT,
                            const int* __restrict__ pos_ids,
                            __bf16* __restrict__ qbf, __bf16* __restrict__ kbf,
                            __bf16* __restrict__ vbf) {
    int idx = blockIdx.x * blockDim.x + threadIdx.x;   // B*Q*48*HD
    int d = idx & (HD - 1);
    int t = idx >> 7;
    int slot = t % 48;
    int row  = t / 48;                 // b*64+q
    int b = row >> 6, q = row & 63;
    const size_t base = (size_t)row * CQKV;
    int pos = pos_ids[b * QLEN + q];
    float c = cosT[(size_t)pos * HD + d];
    float s = sinT[(size_t)pos * HD + d];
    int dp = (d + 64) & 127;           // paired index for rotate_half
    if (slot < NH) {
        float x  = qkv[base + slot*HD + d];
        float xp = qkv[base + slot*HD + dp];
        float rh = (d < 64) ? -xp : xp;
        qbf[(((size_t)b*NH + slot)*QLEN + q)*HD + d] = (__bf16)(x*c + rh*s);
    } else if (slot < NH + NKV) {
        int hk = slot - NH;
        float x  = qkv[base + NH*HD + hk*HD + d];
        float xp = qkv[base + NH*HD + hk*HD + dp];
        float rh = (d < 64) ? -xp : xp;
        kbf[(((size_t)b*NKV + hk)*QLEN + q)*HD + d] = (__bf16)(x*c + rh*s);
    } else {
        int hv = slot - NH - NKV;
        float x = qkv[base + (NH+NKV)*HD + hv*HD + d];
        vbf[(((size_t)b*NKV + hv)*QLEN + q)*HD + d] = (__bf16)x;
    }
}

// =====================================================================
// Kernel 3: flash prefix attention over fp32 KV cache.
// grid (NH, B), 128 threads = 4 waves; wave w owns query rows [w*16, w*16+16).
// K/V chunks (32 tokens x 128d fp32) double-buffered in LDS via
// global_load_async_to_lds_b128 (ASYNCcnt-tracked), overlapping the HBM
// stream of chunk n+1 with the WMMA work on chunk n.
// =====================================================================
__global__ __launch_bounds__(128) void attn_prefix_kernel(
    const __bf16* __restrict__ qbf,
    const float* __restrict__ Kc, const float* __restrict__ Vc,
    const int* __restrict__ cache_lens,
    float* __restrict__ pref_o, float* __restrict__ pref_lse) {
    __shared__ float Kl[2][32][HD + 4];
    __shared__ float Vl[2][32][HD + 4];
    __shared__ float Pl[4][16][32];

    const int qh = blockIdx.x, b = blockIdx.y;
    const int kvh = qh / GRP;
    const int tid = threadIdx.x, wave = tid >> 5, lane = tid & 31;
    const int half = lane >> 4, lm = lane & 15;
    const int cl = cache_lens[b];

    // Q fragments for the wave's 16 rows (4 K-steps of 32 over HD=128)
    v16bf aq[4];
    const __bf16* qrow = qbf + (((size_t)b*NH + qh)*QLEN + wave*16 + lm) * HD;
    #pragma unroll
    for (int t = 0; t < 4; ++t) {
        int db = t*32 + half*16;
        #pragma unroll
        for (int j = 0; j < 8; ++j) { aq[t][j] = qrow[db + j]; aq[t][8+j] = qrow[db + 8 + j]; }
    }

    float mrow[8], lrow[8], alpha[8];
    v8f O[8];
    #pragma unroll
    for (int r = 0; r < 8; ++r) { mrow[r] = NEGINF; lrow[r] = 0.f; }
    #pragma unroll
    for (int dt = 0; dt < 8; ++dt)
        for (int r = 0; r < 8; ++r) O[dt][r] = 0.f;

    // async staging: each thread moves 32 floats of K and 32 of V per chunk
    const int skk = tid >> 2;             // token 0..31
    const int sdp = (tid & 3) * 32;       // d-slice
    auto stageKV = [&](int buf, int l0) {
        const float* kg = Kc + (((size_t)b*LCACHE + l0 + skk)*NKV + kvh)*HD + sdp;
        const float* vg = Vc + (((size_t)b*LCACHE + l0 + skk)*NKV + kvh)*HD + sdp;
        unsigned int kl = lds_lo32(&Kl[buf][skk][sdp]);
        unsigned int vl = lds_lo32(&Vl[buf][skk][sdp]);
        #pragma unroll
        for (int j = 0; j < 32; j += 4) {
            async_ld_b128(kl + j*4, kg + j);
            async_ld_b128(vl + j*4, vg + j);
        }
    };

    const int nch = (cl + 31) >> 5;
    stageKV(0, 0);
    for (int ch = 0; ch < nch; ++ch) {
        const int l0 = ch * 32;
        const int buf = ch & 1;
        if (ch + 1 < nch) { stageKV(buf ^ 1, l0 + 32); WAIT_ASYNC(0x10); }
        else              { WAIT_ASYNC(0x0); }
        __syncthreads();

        // S = Q * K^T for two 16-key tiles
        v8f s0, s1;
        #pragma unroll
        for (int r = 0; r < 8; ++r) { s0[r] = 0.f; s1[r] = 0.f; }
        #pragma unroll
        for (int t = 0; t < 4; ++t) {
            v16bf b0, b1;
            const float* kr0 = &Kl[buf][lm][t*32 + half*16];
            const float* kr1 = &Kl[buf][16 + lm][t*32 + half*16];
            #pragma unroll
            for (int i = 0; i < 16; ++i) { b0[i] = (__bf16)kr0[i]; b1[i] = (__bf16)kr1[i]; }
            s0 = wmma_bf16(aq[t], b0, s0);
            s1 = wmma_bf16(aq[t], b1, s1);
        }

        const bool v0 = (l0 + lm) < cl;
        const bool v1 = (l0 + 16 + lm) < cl;
        #pragma unroll
        for (int r = 0; r < 8; ++r) {
            float x0 = v0 ? s0[r]*SCALE : NEGINF;
            float x1 = v1 ? s1[r]*SCALE : NEGINF;
            float mx = fmaxf(x0, x1);
            for (int off = 1; off < 16; off <<= 1) mx = fmaxf(mx, __shfl_xor(mx, off, 32));
            float mn = fmaxf(mrow[r], mx);
            alpha[r] = __expf(mrow[r] - mn);
            float p0 = v0 ? __expf(x0 - mn) : 0.f;
            float p1 = v1 ? __expf(x1 - mn) : 0.f;
            float rs = p0 + p1;
            for (int off = 1; off < 16; off <<= 1) rs += __shfl_xor(rs, off, 32);
            mrow[r] = mn;
            lrow[r] = lrow[r]*alpha[r] + rs;
            s0[r] = p0; s1[r] = p1;
        }

        // transpose P (C-layout -> A-layout) through per-wave LDS tile
        #pragma unroll
        for (int r = 0; r < 8; ++r) {
            Pl[wave][r + half*8][lm]      = s0[r];
            Pl[wave][r + half*8][16 + lm] = s1[r];
        }
        __syncthreads();
        v16bf ap;
        {
            const float* pr = &Pl[wave][lm][0];
            #pragma unroll
            for (int j = 0; j < 8; ++j) { ap[j] = (__bf16)pr[half*16 + j]; ap[8+j] = (__bf16)pr[half*16 + 8 + j]; }
        }
        // O = O*alpha + P * V
        #pragma unroll
        for (int dt = 0; dt < 8; ++dt) {
            v16bf bv;
            #pragma unroll
            for (int i = 0; i < 16; ++i) bv[i] = (__bf16)Vl[buf][half*16 + i][dt*16 + lm];
            v8f o = O[dt];
            #pragma unroll
            for (int r = 0; r < 8; ++r) o[r] *= alpha[r];
            O[dt] = wmma_bf16(ap, bv, o);
        }
        __syncthreads();
    }

    const size_t obase = ((size_t)b*NH + qh)*QLEN + wave*16;
    #pragma unroll
    for (int r = 0; r < 8; ++r) {
        float inv = 1.0f / lrow[r];
        #pragma unroll
        for (int dt = 0; dt < 8; ++dt)
            pref_o[(obase + r + half*8)*HD + dt*16 + lm] = O[dt][r] * inv;
    }
    if (lm == 0)
        for (int r = 0; r < 8; ++r)
            pref_lse[obase + half*8 + r] = mrow[r] + __logf(lrow[r]);
}

// =====================================================================
// Kernel 4: tree-masked attention over the 64 current tokens (bf16 k/v).
// =====================================================================
__global__ __launch_bounds__(128) void attn_cur_kernel(
    const __bf16* __restrict__ qbf, const __bf16* __restrict__ kbf,
    const __bf16* __restrict__ vbf, const int* __restrict__ tree,
    float* __restrict__ cur_o, float* __restrict__ cur_lse) {
    __shared__ float Pl[4][16][32];
    const int qh = blockIdx.x, b = blockIdx.y;
    const int kvh = qh / GRP;
    const int tid = threadIdx.x, wave = tid >> 5, lane = tid & 31;
    const int half = lane >> 4, lm = lane & 15;

    v16bf aq[4];
    const __bf16* qrow = qbf + (((size_t)b*NH + qh)*QLEN + wave*16 + lm) * HD;
    #pragma unroll
    for (int t = 0; t < 4; ++t) {
        int db = t*32 + half*16;
        #pragma unroll
        for (int j = 0; j < 8; ++j) { aq[t][j] = qrow[db + j]; aq[t][8+j] = qrow[db + 8 + j]; }
    }

    float mrow[8], lrow[8], alpha[8];
    v8f O[8];
    #pragma unroll
    for (int r = 0; r < 8; ++r) { mrow[r] = NEGINF; lrow[r] = 0.f; }
    #pragma unroll
    for (int dt = 0; dt < 8; ++dt)
        for (int r = 0; r < 8; ++r) O[dt][r] = 0.f;

    const __bf16* kbase = kbf + ((size_t)b*NKV + kvh) * QLEN * HD;
    const __bf16* vbase = vbf + ((size_t)b*NKV + kvh) * QLEN * HD;
    const int* trow = tree + (size_t)b * QLEN * QLEN;

    for (int ch = 0; ch < 2; ++ch) {
        const int l0 = ch * 32;
        v8f s0, s1;
        #pragma unroll
        for (int r = 0; r < 8; ++r) { s0[r] = 0.f; s1[r] = 0.f; }
        #pragma unroll
        for (int t = 0; t < 4; ++t) {
            v16bf b0, b1;
            const __bf16* k0p = kbase + (size_t)(l0 + lm)*HD      + t*32 + half*16;
            const __bf16* k1p = kbase + (size_t)(l0 + 16 + lm)*HD + t*32 + half*16;
            #pragma unroll
            for (int i = 0; i < 16; ++i) { b0[i] = k0p[i]; b1[i] = k1p[i]; }
            s0 = wmma_bf16(aq[t], b0, s0);
            s1 = wmma_bf16(aq[t], b1, s1);
        }
        #pragma unroll
        for (int r = 0; r < 8; ++r) {
            int qr = wave*16 + r + half*8;
            bool v0 = trow[(size_t)qr*QLEN + l0 + lm]      != 0;
            bool v1 = trow[(size_t)qr*QLEN + l0 + 16 + lm] != 0;
            float x0 = v0 ? s0[r]*SCALE : NEGINF;
            float x1 = v1 ? s1[r]*SCALE : NEGINF;
            float mx = fmaxf(x0, x1);
            for (int off = 1; off < 16; off <<= 1) mx = fmaxf(mx, __shfl_xor(mx, off, 32));
            float mn = fmaxf(mrow[r], mx);
            alpha[r] = __expf(mrow[r] - mn);
            float p0 = v0 ? __expf(x0 - mn) : 0.f;
            float p1 = v1 ? __expf(x1 - mn) : 0.f;
            float rs = p0 + p1;
            for (int off = 1; off < 16; off <<= 1) rs += __shfl_xor(rs, off, 32);
            mrow[r] = mn;
            lrow[r] = lrow[r]*alpha[r] + rs;
            s0[r] = p0; s1[r] = p1;
        }
        #pragma unroll
        for (int r = 0; r < 8; ++r) {
            Pl[wave][r + half*8][lm]      = s0[r];
            Pl[wave][r + half*8][16 + lm] = s1[r];
        }
        __syncthreads();
        v16bf ap;
        {
            const float* pr = &Pl[wave][lm][0];
            #pragma unroll
            for (int j = 0; j < 8; ++j) { ap[j] = (__bf16)pr[half*16 + j]; ap[8+j] = (__bf16)pr[half*16 + 8 + j]; }
        }
        #pragma unroll
        for (int dt = 0; dt < 8; ++dt) {
            v16bf bv;
            #pragma unroll
            for (int i = 0; i < 16; ++i)
                bv[i] = vbase[(size_t)(l0 + half*16 + i)*HD + dt*16 + lm];
            v8f o = O[dt];
            #pragma unroll
            for (int r = 0; r < 8; ++r) o[r] *= alpha[r];
            O[dt] = wmma_bf16(ap, bv, o);
        }
        __syncthreads();
    }

    const size_t obase = ((size_t)b*NH + qh)*QLEN + wave*16;
    #pragma unroll
    for (int r = 0; r < 8; ++r) {
        float inv = 1.0f / lrow[r];
        #pragma unroll
        for (int dt = 0; dt < 8; ++dt)
            cur_o[(obase + r + half*8)*HD + dt*16 + lm] = O[dt][r] * inv;
    }
    if (lm == 0)
        for (int r = 0; r < 8; ++r)
            cur_lse[obase + half*8 + r] = mrow[r] + __logf(lrow[r]);
}

// =====================================================================
// Kernel 5: LSE-sigmoid merge -> bf16 activations [512 x 4096]
// =====================================================================
__global__ void merge_kernel(const float* __restrict__ pref_o, const float* __restrict__ cur_o,
                             const float* __restrict__ pref_lse, const float* __restrict__ cur_lse,
                             __bf16* __restrict__ attn_bf) {
    int idx = blockIdx.x * blockDim.x + threadIdx.x;  // (((b*NH+qh)*Q+q)*HD+d)
    int d = idx & (HD - 1);
    int t = idx >> 7;
    int q = t & 63; int t2 = t >> 6;
    int qh = t2 & 31; int b = t2 >> 5;
    int li = ((b*NH) + qh) * QLEN + q;
    float w = 1.0f / (1.0f + __expf(-(pref_lse[li] - cur_lse[li])));
    float val = pref_o[idx] * w + cur_o[idx] * (1.0f - w);
    attn_bf[((size_t)(b*QLEN + q)) * (NH*HD) + qh*HD + d] = (__bf16)val;
}

// =====================================================================
// Kernel 6: output projection  out[512 x 4096] = attn_bf * Wo
// =====================================================================
__global__ __launch_bounds__(128) void gemm_out_kernel(
    const __bf16* __restrict__ A, const float* __restrict__ Wo,
    float* __restrict__ out) {
    __shared__ float wtile[2][32][16];
    const int tid  = threadIdx.x;
    const int wave = tid >> 5, lane = tid & 31;
    const int half = lane >> 4, lm = lane & 15;
    const int colBase = blockIdx.x * 16;
    const int rowBase = blockIdx.y * 64 + wave * 16;

    const int e = tid * 4;
    const int skk = e >> 4, snn = e & 15;
    const __bf16* arow = A + (size_t)(rowBase + lm) * HDIM;

    auto stageW = [&](int buf, int k0) {
        const float* wr = Wo + (size_t)(k0 + skk) * HDIM + colBase + snn;
        async_ld_b128(lds_lo32(&wtile[buf][skk][snn]), wr);
    };

    v8f acc; for (int r = 0; r < 8; ++r) acc[r] = 0.f;

    stageW(0, 0);
    for (int k0 = 0; k0 < HDIM; k0 += 32) {
        const int buf = (k0 >> 5) & 1;
        if (k0 + 32 < HDIM) { stageW(buf ^ 1, k0 + 32); WAIT_ASYNC(0x1); }
        else                { WAIT_ASYNC(0x0); }
        __syncthreads();

        const int kb = k0 + half * 8;
        v16bf a, b;
        #pragma unroll
        for (int j = 0; j < 8; ++j) { a[j] = arow[kb + j]; a[8+j] = arow[kb + 16 + j]; }
        #pragma unroll
        for (int i = 0; i < 16; ++i) b[i] = (__bf16)wtile[buf][(i < 8 ? half*8 + i : 8 + half*8 + i)][lm];
        acc = wmma_bf16(a, b, acc);
        __syncthreads();
    }
    #pragma unroll
    for (int r = 0; r < 8; ++r)
        out[(size_t)(rowBase + r + half*8) * HDIM + colBase + lm] = acc[r];
}

// =====================================================================
// Host launcher
// =====================================================================
extern "C" void kernel_launch(void* const* d_in, const int* in_sizes, int n_in,
                              void* d_out, int out_size, void* d_ws, size_t ws_size,
                              hipStream_t stream) {
    const float* X    = (const float*)d_in[0];
    const float* Wq   = (const float*)d_in[1];
    const float* Wk   = (const float*)d_in[2];
    const float* Wv   = (const float*)d_in[3];
    const float* Wo   = (const float*)d_in[4];
    const float* Kc   = (const float*)d_in[5];
    const float* Vc   = (const float*)d_in[6];
    const float* cosT = (const float*)d_in[7];
    const float* sinT = (const float*)d_in[8];
    const int*   tree = (const int*)d_in[9];
    const int*   pos  = (const int*)d_in[10];
    const int*   clen = (const int*)d_in[11];

    char* ws = (char*)d_ws;
    float*  qkv_raw = (float*)ws;   ws += (size_t)BATCH*QLEN*CQKV*sizeof(float);
    __bf16* qbf     = (__bf16*)ws;  ws += (size_t)BATCH*NH*QLEN*HD*sizeof(__bf16);
    __bf16* kbf     = (__bf16*)ws;  ws += (size_t)BATCH*NKV*QLEN*HD*sizeof(__bf16);
    __bf16* vbf     = (__bf16*)ws;  ws += (size_t)BATCH*NKV*QLEN*HD*sizeof(__bf16);
    float*  pref_o  = (float*)ws;   ws += (size_t)BATCH*NH*QLEN*HD*sizeof(float);
    float*  cur_o   = (float*)ws;   ws += (size_t)BATCH*NH*QLEN*HD*sizeof(float);
    float*  pref_lse= (float*)ws;   ws += (size_t)BATCH*NH*QLEN*sizeof(float);
    float*  cur_lse = (float*)ws;   ws += (size_t)BATCH*NH*QLEN*sizeof(float);
    __bf16* attn_bf = (__bf16*)ws;  ws += (size_t)BATCH*QLEN*NH*HD*sizeof(__bf16);

    gemm_qkv_kernel<<<dim3(CQKV/16, (BATCH*QLEN)/64), 128, 0, stream>>>(X, Wq, Wk, Wv, qkv_raw);
    rope_kernel<<<(BATCH*QLEN*48*HD)/256, 256, 0, stream>>>(qkv_raw, cosT, sinT, pos, qbf, kbf, vbf);
    attn_prefix_kernel<<<dim3(NH, BATCH), 128, 0, stream>>>(qbf, Kc, Vc, clen, pref_o, pref_lse);
    attn_cur_kernel<<<dim3(NH, BATCH), 128, 0, stream>>>(qbf, kbf, vbf, tree, cur_o, cur_lse);
    merge_kernel<<<(BATCH*QLEN*NH*HD)/256, 256, 0, stream>>>(pref_o, cur_o, pref_lse, cur_lse, attn_bf);
    gemm_out_kernel<<<dim3(HDIM/16, (BATCH*QLEN)/64), 128, 0, stream>>>(attn_bf, Wo, (float*)d_out);
}
